// Cross_GNN_31413390803490
// MI455X (gfx1250) — matrix operand order repaired and checked by reading, not compiled, not run on hardware
//
#include <hip/hip_runtime.h>
#include <hip/hip_bf16.h>
#include <math.h>

#define N_NODES 50000
#define N_INPUT 256
#define HID 32
#define OUTF 128
#define EPSV 1e-12f

typedef __attribute__((ext_vector_type(16))) __bf16 v16bf;
typedef __attribute__((ext_vector_type(8)))  float  v8f;

// ---------------------------------------------------------------- utilities
__global__ void fill_f32(float* __restrict__ p, float v, int n) {
    int i = blockIdx.x * blockDim.x + threadIdx.x;
    if (i < n) p[i] = v;
}

__global__ void deg_count(const int* __restrict__ dst, float* __restrict__ deg, int E) {
    int i = blockIdx.x * blockDim.x + threadIdx.x;
    if (i < E) atomicAdd(&deg[dst[i]], 1.0f);
}

__global__ void deg_to_dinv(float* __restrict__ d, int n) {
    int i = blockIdx.x * blockDim.x + threadIdx.x;
    if (i < n) d[i] = rsqrtf(fmaxf(d[i], 1.0f));
}

// ------------------------------------------------- B pre-pack (f32 -> bf16 frags)
// Pack B[K x NC] into WMMA B-fragment order: fragment (kc, nt) = 32 lanes x 16
// bf16, contiguous. Element j of lane l <- B[(kc*32 + (j<8 ? 8*(l>>4) + j : 16 + 8*(l>>4) + j-8))*NC + nt*16 + (l&15)].
__global__ void pack_B_bf16(const float* __restrict__ B, __bf16* __restrict__ P,
                            int K, int NC) {
    int nt   = blockIdx.x;           // n-tile (16 cols)
    int kc   = blockIdx.y;           // k-chunk (32 rows)
    int lane = threadIdx.x & 31;     // blockDim.x == 32
    int half = lane >> 4;
    int lr   = lane & 15;
    __bf16* dstp = P + (((size_t)kc * (NC / 16) + nt) * 32 + lane) * 16;
#pragma unroll
    for (int j = 0; j < 16; ++j) {
        int kb = kc * 32 + (j < 8 ? 8 * half + j : 16 + 8 * half + (j - 8));
        dstp[j] = (__bf16)B[kb * NC + nt * 16 + lr];
    }
}

// ---------------------------------------------------------------- WMMA GEMM
// C[M x (NTILES*16)] = A[M x K] (f32) * B (pre-packed bf16 fragments).
// One wave computes a full 16-row x NC strip: the A fragment is loaded once per
// K-chunk (4 x b128) and fed to NTILES back-to-back v_wmma ops. M % 16 == 0.
// blockDim.x = 256 (8 waves, 8 strips per block).
template <int NTILES>
__global__ void gemm_wmma_bf16(const float* __restrict__ A,
                               const __bf16* __restrict__ PB,
                               float* __restrict__ C,
                               int M, int K, int ldc) {
    const int lane = threadIdx.x & 31;
    const int wave = threadIdx.x >> 5;
    const int m0   = (blockIdx.x * 8 + wave) * 16;
    if (m0 >= M) return;                          // idle trailing waves only

    const int half = lane >> 4;
    const int lr   = lane & 15;
    const unsigned rowBase = (unsigned)(m0 + lr) * (unsigned)K;  // fits 32-bit

    v8f acc[NTILES];
#pragma unroll
    for (int t = 0; t < NTILES; ++t)
        acc[t] = (v8f){0.f, 0.f, 0.f, 0.f, 0.f, 0.f, 0.f, 0.f};

    for (int k0 = 0; k0 < K; k0 += 32) {
        // ---- A fragment: two contiguous 8-float runs per lane (4 x b128)
        const float4* arow = (const float4*)(A + rowBase + k0);
        float4 a0 = arow[2 * half + 0];           // K = 8*half + 0..3
        float4 a1 = arow[2 * half + 1];           // K = 8*half + 4..7
        float4 a2 = arow[4 + 2 * half + 0];       // K = 16 + 8*half + 0..3
        float4 a3 = arow[4 + 2 * half + 1];       // K = 16 + 8*half + 4..7
        v16bf af;
        af[0]  = (__bf16)a0.x; af[1]  = (__bf16)a0.y;
        af[2]  = (__bf16)a0.z; af[3]  = (__bf16)a0.w;
        af[4]  = (__bf16)a1.x; af[5]  = (__bf16)a1.y;
        af[6]  = (__bf16)a1.z; af[7]  = (__bf16)a1.w;
        af[8]  = (__bf16)a2.x; af[9]  = (__bf16)a2.y;
        af[10] = (__bf16)a2.z; af[11] = (__bf16)a2.w;
        af[12] = (__bf16)a3.x; af[13] = (__bf16)a3.y;
        af[14] = (__bf16)a3.z; af[15] = (__bf16)a3.w;

        // ---- NTILES B fragments (contiguous in packed buffer), one wmma each
        const __bf16* pbase = PB + (((unsigned)(k0 >> 5) * NTILES * 32u + (unsigned)lane) * 16u);
#pragma unroll
        for (int t = 0; t < NTILES; ++t) {
            v16bf bfv = *(const v16bf*)(pbase + t * 512);  // 32B vector load
            acc[t] = __builtin_amdgcn_wmma_f32_16x16x32_bf16(
                /*neg_a=*/false, af, /*neg_b=*/false, bfv,
                /*c_mod=*/(short)0, acc[t], /*reuse_a=*/false, /*reuse_b=*/false);
        }
    }

    // C/D layout: VGPR r -> M = r + 8*half, N = lr (+16 per tile)
#pragma unroll
    for (int t = 0; t < NTILES; ++t) {
#pragma unroll
        for (int r = 0; r < 8; ++r) {
            C[(unsigned)(m0 + r + 8 * half) * (unsigned)ldc + t * 16 + lr] = acc[t][r];
        }
    }
}

// ------------------------------------------------- GCN scatter (normalized A+I)
// out[i, f] = bias[f] + dinv[i]^2 * h[i, f]   (self-loop term, initializes out)
template <int FSHIFT>
__global__ void self_init(const float* __restrict__ h, const float* __restrict__ dinv,
                          const float* __restrict__ bias, float* __restrict__ out,
                          int N, int ldh, int ldo, int coff) {
    const int F = 1 << FSHIFT;
    int i = blockIdx.x * blockDim.x + threadIdx.x;
    if (i >= (N << FSHIFT)) return;
    int node = i >> FSHIFT;
    int f    = i & (F - 1);
    float dv = dinv[node];
    out[node * ldo + coff + f] = bias[f] + h[node * ldh + f] * dv * dv;
}

// out[dst, f] += dinv[src]*dinv[dst] * h[src, f]; one lane per (edge, feature):
// each wave does a coalesced gather of h[src] and F coalesced L2 f32 atomics.
template <int FSHIFT>
__global__ void edge_scatter(const int* __restrict__ src, const int* __restrict__ dst,
                             const float* __restrict__ h, const float* __restrict__ dinv,
                             float* __restrict__ out,
                             int E, int ldh, int ldo, int coff) {
    const int F = 1 << FSHIFT;
    int i = blockIdx.x * blockDim.x + threadIdx.x;
    if (i >= (E << FSHIFT)) return;
    int e = i >> FSHIFT;
    int f = i & (F - 1);
    int s = src[e];
    int d = dst[e];
    float norm = dinv[s] * dinv[d];
    atomicAdd(&out[d * ldo + coff + f], h[s * ldh + f] * norm);
}

// ------------------------------------------------- cross-view cosine mixing
// one wave32 per node (32 lanes == HID features); butterfly shuffle reductions.
__global__ void sim_combine(const float* __restrict__ x1, const float* __restrict__ x2,
                            float* __restrict__ mian, float* __restrict__ sup, int N) {
    int lane = threadIdx.x & 31;
    int node = (blockIdx.x * blockDim.x + threadIdx.x) >> 5;
    if (node >= N) return;
    int idx = node * HID + lane;
    float a = x1[idx], b = x2[idx];
    float na = a * a, nb = b * b, ab = a * b;
#pragma unroll
    for (int off = 16; off > 0; off >>= 1) {
        na += __shfl_xor(na, off, 32);
        nb += __shfl_xor(nb, off, 32);
        ab += __shfl_xor(ab, off, 32);
    }
    float sim = ab / (fmaxf(sqrtf(na), EPSV) * fmaxf(sqrtf(nb), EPSV));
    mian[idx] = a + b * sim;
    sup[idx]  = b + a * sim;
}

// ---------------------------------------------------------------- driver
extern "C" void kernel_launch(void* const* d_in, const int* in_sizes, int n_in,
                              void* d_out, int out_size, void* d_ws, size_t ws_size,
                              hipStream_t stream) {
    const float* x     = (const float*)d_in[0];
    const int*   ei_u  = (const int*)d_in[1];
    const int*   ei_u2 = (const int*)d_in[2];
    const float* W_in  = (const float*)d_in[3];
    const float* b_in  = (const float*)d_in[4];
    const float* W_hid = (const float*)d_in[5];
    const float* b_hid = (const float*)d_in[6];
    const float* W_out = (const float*)d_in[7];
    const float* b_out = (const float*)d_in[8];
    float* out = (float*)d_out;

    const int N = N_NODES;
    const int E = in_sizes[1] / 2;
    const int* src_u  = ei_u;       const int* dst_u  = ei_u + E;
    const int* src_u2 = ei_u2;      const int* dst_u2 = ei_u2 + E;

    // workspace carve-out (~84 MB of f32 + ~26 KB packed weights)
    float* ws = (float*)d_ws;
    float* dinv_u  = ws; ws += N;
    float* dinv_u2 = ws; ws += N;
    float* h   = ws; ws += (size_t)N * HID;   // shared GEMM result / layer-2 h1
    float* x1  = ws; ws += (size_t)N * HID;
    float* x2  = ws; ws += (size_t)N * HID;
    float* mn  = ws; ws += (size_t)N * HID;
    float* sp  = ws; ws += (size_t)N * HID;
    float* g1  = ws; ws += (size_t)N * OUTF;  // layer-2 h2 (HID cols), layer-3 h1
    float* g2  = ws; ws += (size_t)N * OUTF;  // layer-3 h2
    __bf16* pb_in  = (__bf16*)ws;             // 8 kchunks * 2 ntiles * 512
    __bf16* pb_hid = pb_in  + (N_INPUT / 32) * (HID / 16) * 512;
    __bf16* pb_out = pb_hid + (HID / 32) * (HID / 16) * 512;
    (void)ws_size; (void)n_in; (void)out_size;

    const int TB = 256;
    dim3 blk(TB);
    auto blocks = [](long long n) { return dim3((unsigned)((n + 255) / 256)); };

    // ---- degree -> D^{-1/2} for both edge sets (self-loop = init 1.0)
    fill_f32<<<blocks(N), blk, 0, stream>>>(dinv_u, 1.0f, N);
    fill_f32<<<blocks(N), blk, 0, stream>>>(dinv_u2, 1.0f, N);
    deg_count<<<blocks(E), blk, 0, stream>>>(dst_u,  dinv_u,  E);
    deg_count<<<blocks(E), blk, 0, stream>>>(dst_u2, dinv_u2, E);
    deg_to_dinv<<<blocks(N), blk, 0, stream>>>(dinv_u,  N);
    deg_to_dinv<<<blocks(N), blk, 0, stream>>>(dinv_u2, N);

    // ---- pre-pack the three weight matrices into bf16 WMMA B-fragments
    pack_B_bf16<<<dim3(HID / 16, N_INPUT / 32), dim3(32), 0, stream>>>(W_in,  pb_in,  N_INPUT, HID);
    pack_B_bf16<<<dim3(HID / 16,  HID / 32),    dim3(32), 0, stream>>>(W_hid, pb_hid, HID, HID);
    pack_B_bf16<<<dim3(OUTF / 16, HID / 32),    dim3(32), 0, stream>>>(W_out, pb_out, HID, OUTF);

    const int mtiles = N / 16;                 // 3125, exact
    dim3 gemm_grid((mtiles + 7) / 8);

    // ---- layer 1: h = x @ W_in (shared by both views)
    gemm_wmma_bf16<HID / 16><<<gemm_grid, blk, 0, stream>>>(x, pb_in, h, N, N_INPUT, HID);
    // x1 = Au h + b, x2 = Au2 h + b
    self_init<5><<<blocks((long long)N * HID), blk, 0, stream>>>(h, dinv_u,  b_in, x1, N, HID, HID, 0);
    self_init<5><<<blocks((long long)N * HID), blk, 0, stream>>>(h, dinv_u2, b_in, x2, N, HID, HID, 0);
    edge_scatter<5><<<blocks((long long)E * HID), blk, 0, stream>>>(src_u,  dst_u,  h, dinv_u,  x1, E, HID, HID, 0);
    edge_scatter<5><<<blocks((long long)E * HID), blk, 0, stream>>>(src_u2, dst_u2, h, dinv_u2, x2, E, HID, HID, 0);
    sim_combine<<<blocks((long long)N * 32), blk, 0, stream>>>(x1, x2, mn, sp, N);

    // ---- layer 2: h1 = mn @ W_hid, h2 = sp @ W_hid
    gemm_wmma_bf16<HID / 16><<<gemm_grid, blk, 0, stream>>>(mn, pb_hid, h,  N, HID, HID);
    gemm_wmma_bf16<HID / 16><<<gemm_grid, blk, 0, stream>>>(sp, pb_hid, g1, N, HID, HID);
    self_init<5><<<blocks((long long)N * HID), blk, 0, stream>>>(h,  dinv_u,  b_hid, x1, N, HID, HID, 0);
    self_init<5><<<blocks((long long)N * HID), blk, 0, stream>>>(g1, dinv_u2, b_hid, x2, N, HID, HID, 0);
    edge_scatter<5><<<blocks((long long)E * HID), blk, 0, stream>>>(src_u,  dst_u,  h,  dinv_u,  x1, E, HID, HID, 0);
    edge_scatter<5><<<blocks((long long)E * HID), blk, 0, stream>>>(src_u2, dst_u2, g1, dinv_u2, x2, E, HID, HID, 0);
    sim_combine<<<blocks((long long)N * 32), blk, 0, stream>>>(x1, x2, mn, sp, N);

    // ---- layer 3: both branches use edge_u (as in reference source)
    gemm_wmma_bf16<OUTF / 16><<<gemm_grid, blk, 0, stream>>>(mn, pb_out, g1, N, HID, OUTF);
    gemm_wmma_bf16<OUTF / 16><<<gemm_grid, blk, 0, stream>>>(sp, pb_out, g2, N, HID, OUTF);
    // out[:, 0:128] = Au g1 + b_out ; out[:, 128:256] = Au g2 + b_out
    self_init<7><<<blocks((long long)N * OUTF), blk, 0, stream>>>(g1, dinv_u, b_out, out, N, OUTF, 2 * OUTF, 0);
    self_init<7><<<blocks((long long)N * OUTF), blk, 0, stream>>>(g2, dinv_u, b_out, out, N, OUTF, 2 * OUTF, OUTF);
    edge_scatter<7><<<blocks((long long)E * OUTF), blk, 0, stream>>>(src_u, dst_u, g1, dinv_u, out, E, OUTF, 2 * OUTF, 0);
    edge_scatter<7><<<blocks((long long)E * OUTF), blk, 0, stream>>>(src_u, dst_u, g2, dinv_u, out, E, OUTF, 2 * OUTF, OUTF);
}
